// MRiemGNNLayer_14370960572833
// MI455X (gfx1250) — matrix operand order repaired and checked by reading, not compiled, not run on hardware
//
#include <hip/hip_runtime.h>
#include <stdint.h>

#define N_NODES 100000
#define IN_DIM  256
#define OUT_DIM 256
#define K_RFF   64
#define R_REL   3
#define S_DIM   2
#define E_EDGES 1600000
#define NODE_TILES (N_NODES / 16)   // 6250 exactly
#define TPW 5                       // node tiles per wave (GEMM kernels)

typedef __attribute__((ext_vector_type(16))) _Float16 v16h;
typedef __attribute__((ext_vector_type(8)))  _Float16 v8h;
typedef __attribute__((ext_vector_type(2)))  _Float16 v2h;
typedef __attribute__((ext_vector_type(8)))  float    v8f;

static __device__ __forceinline__ v8f wmma_f16(v16h a, v16h b, v8f c) {
  // D = A(16x32 f16) x B(32x16 f16) + C(16x16 f32)
  return __builtin_amdgcn_wmma_f32_16x16x32_f16(false, a, false, b, (short)0, c,
                                                false, false);
}

// Build A-fragment (16x32 f16) from a transposed [col][inner] f16 buffer:
// two contiguous 16B reads -> ds_load_b128 / global_load_b128.
static __device__ __forceinline__ v16h frag_a(const _Float16* base) {
  v8h lo = *(const v8h*)base;
  v8h hi = *(const v8h*)(base + 8);
  return __builtin_shufflevector(lo, hi, 0, 1, 2, 3, 4, 5, 6, 7,
                                         8, 9, 10, 11, 12, 13, 14, 15);
}

static __device__ __forceinline__ void atomic_pk_add_f16(_Float16* p, v2h v) {
#if __has_builtin(__builtin_amdgcn_global_atomic_fadd_v2f16)
  __builtin_amdgcn_global_atomic_fadd_v2f16(
      (__attribute__((address_space(1))) v2h*)(uintptr_t)p, v);
#else
  asm volatile("global_atomic_pk_add_f16 %0, %1, off"
               :: "v"(p), "v"(v) : "memory");
#endif
}

// ---------------------------------------------------------------------------
// Kernel 1: phi[s,r,n,k] = scale * cos( (h @ w_rff[s,r])[n,k] + b_rff[s,r,k] )
// D[M=k_out(4x16 tiles), N=node(16)]: each lane stores 8 contiguous f16 (16B).
// w_rff[s,r] staged transposed into LDS as f16 [k_out][i] (32 KB).
// ---------------------------------------------------------------------------
__global__ __launch_bounds__(256) void k_phi(const float* __restrict__ h,
                                             const float* __restrict__ w_rff,
                                             const float* __restrict__ b_rff,
                                             _Float16* __restrict__ phi) {
  const int sr   = blockIdx.y;          // 0..5  (s*R + r)
  const int lane = threadIdx.x & 31;
  const int wave = threadIdx.x >> 5;
  const int nlo  = lane & 15;
  const int hi   = lane >> 4;
  const int koff  = hi * 16;
  const int kbase = hi * 16;

  __shared__ _Float16 sWt[K_RFF * IN_DIM];        // transposed, 32 KB
  const float* wsrc = w_rff + (size_t)sr * IN_DIM * K_RFF;
  for (int idx = threadIdx.x; idx < IN_DIM * K_RFF; idx += 256) {
    const int i = idx >> 6;            // inner (0..255)
    const int k = idx & 63;            // k_out
    sWt[k * IN_DIM + i] = (_Float16)wsrc[idx];    // coalesced global read
  }
  __syncthreads();

  const float scale = 0.17677669529663687f;       // sqrt(2/64), SIGMA = 1
  const float* bp = b_rff + (size_t)sr * K_RFF;

  for (int t = 0; t < TPW; ++t) {
    const int tile = (blockIdx.x * 8 + wave) * TPW + t;
    if (tile >= NODE_TILES) break;
    const int node = tile * 16 + nlo;

    v8f acc[4] = {};
    for (int ii = 0; ii < IN_DIM; ii += 32) {
      v16h b;
      const float* hp = h + (size_t)node * IN_DIM + ii + kbase;
#pragma unroll
      for (int j = 0; j < 16; ++j) b[j] = (_Float16)hp[j];
#pragma unroll
      for (int mt = 0; mt < 4; ++mt) {
        const int mcol = mt * 16 + nlo;
        const v16h a = frag_a(&sWt[mcol * IN_DIM + ii + koff]);
        acc[mt] = wmma_f16(a, b, acc[mt]);
      }
    }

#pragma unroll
    for (int mt = 0; mt < 4; ++mt) {
      _Float16 o8[8];
#pragma unroll
      for (int v = 0; v < 8; ++v) {
        const int kout = mt * 16 + hi * 8 + v;    // D row M = v + 8*hi
        o8[v] = (_Float16)(__cosf(acc[mt][v] + bp[kout]) * scale);
      }
      float4* dst = (float4*)(phi + ((size_t)sr * N_NODES + node) * K_RFF +
                              mt * 16 + hi * 8);
      *dst = *(const float4*)o8;
    }
  }
}

// ---------------------------------------------------------------------------
// Kernel 2: edge gather/scatter + degree. One wave per edge: coalesced 128B
// f16 gather of phi[s,r,col,:], 32 packed-f16 atomics into agg[s,r,row,:]
// (phi+agg = 153.6 MB -> L2-resident). Lane 0 bumps deg.
// ---------------------------------------------------------------------------
__global__ __launch_bounds__(256) void k_scatter(const int* __restrict__ edges,
                                                 const _Float16* __restrict__ phi,
                                                 _Float16* __restrict__ agg,
                                                 float* __restrict__ deg) {
  const int r      = blockIdx.y;
  const int lane   = threadIdx.x & 31;
  const int gwave  = (int)((blockIdx.x * blockDim.x + threadIdx.x) >> 5);
  const int nwaves = (int)((gridDim.x * blockDim.x) >> 5);
  const int* rowp = edges + (size_t)(r * 2 + 0) * E_EDGES;
  const int* colp = edges + (size_t)(r * 2 + 1) * E_EDGES;

  for (int e = gwave; e < E_EDGES; e += nwaves) {
    const int row = rowp[e];
    const int col = colp[e];
    if (lane == 0) atomicAdd(&deg[(size_t)r * N_NODES + row], 1.0f);
#pragma unroll
    for (int s = 0; s < S_DIM; ++s) {
      const v2h* p = (const v2h*)(phi +
          ((size_t)(s * R_REL + r) * N_NODES + col) * K_RFF);
      const v2h val = p[lane];
      _Float16* ap = agg + ((size_t)(s * R_REL + r) * N_NODES + row) * K_RFF +
                     2 * lane;
      atomic_pk_add_f16(ap, val);
    }
  }
}

// ---------------------------------------------------------------------------
// Kernel 3: out[s,n,o] = sum_r alpha[s,r]/deg[r,n] * (agg[s,r,n,:] @ W[s,r])
//                        + (h[n,:] @ self_loop[s]) + bias[s,o]
// D[M=out(4x16 tiles), N=node(16)]. self_loop/W slices staged transposed in
// LDS (56 KB); alpha/deg folded into the f16 B operand.
// ---------------------------------------------------------------------------
__global__ __launch_bounds__(256) void k_out(const float* __restrict__ h,
                                             const _Float16* __restrict__ agg,
                                             const float* __restrict__ deg,
                                             const float* __restrict__ W,
                                             const float* __restrict__ self_loop,
                                             const float* __restrict__ bias,
                                             const float* __restrict__ rel_w,
                                             float* __restrict__ out) {
  const int s    = blockIdx.z;
  const int og   = blockIdx.y;          // 4 groups of 64 outputs
  const int lane = threadIdx.x & 31;
  const int wave = threadIdx.x >> 5;
  const int nlo  = lane & 15;
  const int hi   = lane >> 4;
  const int koff  = hi * 16;
  const int kbase = hi * 16;

  __shared__ _Float16 sSelf[64 * IN_DIM];          // [ocol][i]   32 KB
  __shared__ _Float16 sWr[R_REL * 64 * K_RFF];     // [r][ocol][k] 24 KB

  for (int idx = threadIdx.x; idx < 64 * IN_DIM; idx += 256) {
    const int i = idx >> 6, c = idx & 63;
    sSelf[c * IN_DIM + i] =
        (_Float16)self_loop[((size_t)s * IN_DIM + i) * OUT_DIM + og * 64 + c];
  }
  for (int idx = threadIdx.x; idx < R_REL * 64 * K_RFF; idx += 256) {
    const int r = idx >> 12, rem = idx & 4095;
    const int k = rem >> 6, c = rem & 63;
    sWr[(r * 64 + c) * K_RFF + k] =
        (_Float16)W[((size_t)(s * R_REL + r) * K_RFF + k) * OUT_DIM + og * 64 + c];
  }
  __syncthreads();

  // alpha = softmax(rel_w[s, :])
  const float a0 = rel_w[s * R_REL + 0];
  const float a1 = rel_w[s * R_REL + 1];
  const float a2 = rel_w[s * R_REL + 2];
  const float mx = fmaxf(a0, fmaxf(a1, a2));
  const float e0 = __expf(a0 - mx), e1 = __expf(a1 - mx), e2 = __expf(a2 - mx);
  const float inv = 1.0f / (e0 + e1 + e2);
  const float alpha[R_REL] = {e0 * inv, e1 * inv, e2 * inv};

  for (int t = 0; t < TPW; ++t) {
    const int tile = (blockIdx.x * 8 + wave) * TPW + t;
    if (tile >= NODE_TILES) break;
    const int node = tile * 16 + nlo;

    v8f acc[4] = {};

    // Self-loop GEMM: inner = IN_DIM
    for (int ii = 0; ii < IN_DIM; ii += 32) {
      v16h b;
      const float* hp = h + (size_t)node * IN_DIM + ii + kbase;
#pragma unroll
      for (int j = 0; j < 16; ++j) b[j] = (_Float16)hp[j];
#pragma unroll
      for (int mt = 0; mt < 4; ++mt) {
        const v16h a = frag_a(&sSelf[(mt * 16 + nlo) * IN_DIM + ii + koff]);
        acc[mt] = wmma_f16(a, b, acc[mt]);
      }
    }

    // Relation GEMMs: inner = K_RFF; scale f16 agg by alpha/deg on the fly.
#pragma unroll
    for (int r = 0; r < R_REL; ++r) {
      const float d  = deg[(size_t)r * N_NODES + node];
      const float sc = alpha[r] / fmaxf(d, 1.0f);
      const _Float16* ag = agg +
          ((size_t)(s * R_REL + r) * N_NODES + node) * K_RFF;
#pragma unroll
      for (int kk = 0; kk < K_RFF; kk += 32) {
        const v8h r0 = *(const v8h*)(ag + kk + kbase);
        const v8h r1 = *(const v8h*)(ag + kk + kbase + 8);
        v16h b;
#pragma unroll
        for (int j = 0; j < 8; ++j) {
          b[j]     = (_Float16)((float)r0[j] * sc);
          b[8 + j] = (_Float16)((float)r1[j] * sc);
        }
#pragma unroll
        for (int mt = 0; mt < 4; ++mt) {
          const v16h a =
              frag_a(&sWr[(r * 64 + mt * 16 + nlo) * K_RFF + kk + koff]);
          acc[mt] = wmma_f16(a, b, acc[mt]);
        }
      }
    }

    // Epilogue: + bias, 8 contiguous f32 per tile (two 16B stores).
#pragma unroll
    for (int mt = 0; mt < 4; ++mt) {
      float o8[8];
#pragma unroll
      for (int v = 0; v < 8; ++v) {
        const int o = og * 64 + mt * 16 + hi * 8 + v;
        o8[v] = acc[mt][v] + bias[s * OUT_DIM + o];
      }
      float* op = out + ((size_t)s * N_NODES + node) * OUT_DIM +
                  og * 64 + mt * 16 + hi * 8;
      *(float4*)op       = *(const float4*)o8;
      *(float4*)(op + 4) = *(const float4*)(o8 + 4);
    }
  }
}

// ---------------------------------------------------------------------------
extern "C" void kernel_launch(void* const* d_in, const int* in_sizes, int n_in,
                              void* d_out, int out_size, void* d_ws, size_t ws_size,
                              hipStream_t stream) {
  (void)in_sizes; (void)n_in; (void)out_size; (void)ws_size;

  const float* h         = (const float*)d_in[0];
  const int*   edges     = (const int*)  d_in[1];
  const float* w_rff     = (const float*)d_in[2];
  const float* b_rff     = (const float*)d_in[3];
  const float* W         = (const float*)d_in[4];
  const float* self_loop = (const float*)d_in[5];
  const float* bias      = (const float*)d_in[6];
  const float* rel_w     = (const float*)d_in[7];
  float*       out       = (float*)d_out;

  char* ws = (char*)d_ws;
  const size_t phiBytes = (size_t)S_DIM * R_REL * N_NODES * K_RFF * sizeof(_Float16); // 76.8 MB
  const size_t aggBytes = (size_t)S_DIM * R_REL * N_NODES * K_RFF * sizeof(_Float16); // 76.8 MB
  const size_t degBytes = (size_t)R_REL * N_NODES * sizeof(float);                    // 1.2 MB
  _Float16* phi = (_Float16*)ws;
  _Float16* agg = (_Float16*)(ws + phiBytes);
  float*    deg = (float*)(ws + phiBytes + aggBytes);

  hipMemsetAsync(agg, 0, aggBytes, stream);
  hipMemsetAsync(deg, 0, degBytes, stream);

  const unsigned gemm_bx = (NODE_TILES + 8 * TPW - 1) / (8 * TPW);   // 157
  {
    dim3 grid(gemm_bx, S_DIM * R_REL);
    k_phi<<<grid, 256, 0, stream>>>(h, w_rff, b_rff, phi);
  }
  {
    dim3 grid(2048, R_REL);   // 16384 waves/relation, ~98 edges each
    k_scatter<<<grid, 256, 0, stream>>>(edges, phi, agg, deg);
  }
  {
    dim3 grid(gemm_bx, 4, S_DIM);
    k_out<<<grid, 256, 0, stream>>>(h, agg, deg, W, self_loop, bias, rel_w, out);
  }
}
